// SelfAttention2d_16638703305249
// MI455X (gfx1250) — matrix operand order repaired and checked by reading, not compiled, not run on hardware
//
#include <hip/hip_runtime.h>

#define NB  8
#define NC  256
#define NN  4096
#define NCQ 32
#define XPAD 272   // halfs per LDS x row: 544 B = 16B-aligned stride

typedef __attribute__((ext_vector_type(16))) _Float16 v16h;
typedef __attribute__((ext_vector_type(8)))  _Float16 v8h;
typedef __attribute__((ext_vector_type(8)))  float    v8f;

union V16U { v16h v; v8h h[2]; };

__device__ inline v8f wmma_f16(v16h a, v16h b, v8f c) {
    // (neg_a, A, neg_b, B, c_mod, C, reuse_a, reuse_b)
    return __builtin_amdgcn_wmma_f32_16x16x32_f16(false, a, false, b,
                                                  (short)0, c, false, false);
}

// max over the 16 lanes of each half-wave row, pure VALU DPP (no LDS):
// quad_perm xor1, quad_perm xor2, row_half_mirror, row_mirror
__device__ inline float rowmax16(float v) {
    int t;
    t = __builtin_amdgcn_update_dpp(0, __builtin_bit_cast(int, v), 0x0B1, 0xF, 0xF, true);
    v = fmaxf(v, __builtin_bit_cast(float, t));
    t = __builtin_amdgcn_update_dpp(0, __builtin_bit_cast(int, v), 0x04E, 0xF, 0xF, true);
    v = fmaxf(v, __builtin_bit_cast(float, t));
    t = __builtin_amdgcn_update_dpp(0, __builtin_bit_cast(int, v), 0x141, 0xF, 0xF, true);
    v = fmaxf(v, __builtin_bit_cast(float, t));
    t = __builtin_amdgcn_update_dpp(0, __builtin_bit_cast(int, v), 0x140, 0xF, 0xF, true);
    v = fmaxf(v, __builtin_bit_cast(float, t));
    return v;
}

// ---------------------------------------------------------------------------
// Convert weights to f16 once (wv 256x256, wq/wk 32x256, row-major kept).
// ---------------------------------------------------------------------------
__global__ __launch_bounds__(256) void cvt_w_kernel(
    const float* __restrict__ wv, const float* __restrict__ wq,
    const float* __restrict__ wk,
    _Float16* __restrict__ wvh, _Float16* __restrict__ wqh,
    _Float16* __restrict__ wkh)
{
    const int i = blockIdx.x * 256 + threadIdx.x;
    if (i < NC * NC)  wvh[i] = (_Float16)wv[i];
    if (i < NCQ * NC) { wqh[i] = (_Float16)wq[i]; wkh[i] = (_Float16)wk[i]; }
}

// Stage x[b, :, n0:n0+64] into LDS as f16 [n][cc] (transposed, coalesced reads).
__device__ inline void stage_x_tile(_Float16* lds, const float* __restrict__ x,
                                    int b, int n0, int tid)
{
    const float* xb = x + ((long long)b * NC) * NN + n0;
    for (int i = tid; i < NC * 64; i += 256) {
        const int nn = i & 63;
        const int cc = i >> 6;
        lds[nn * XPAD + cc] = (_Float16)xb[(long long)cc * NN + nn];
    }
}

// ---------------------------------------------------------------------------
// V projection as WMMA GEMM: v[b,c,n] = wv[c,:]·x[b,:,n] + bv[c]  (f16 [B,C,N])
// block = 256 thr = 8 waves, covers (b, 64-n chunk) x all 256 channels.
// wave w owns c-tiles {w, w+8}, 4 n-subtiles each -> 8 accumulators.
// x read from HBM exactly once over the whole kernel.
// ---------------------------------------------------------------------------
__global__ __launch_bounds__(256) void proj_v_kernel(
    const float* __restrict__ x, const _Float16* __restrict__ wvh,
    const float* __restrict__ bv, _Float16* __restrict__ vh)
{
    __shared__ __align__(16) _Float16 ldsX[64 * XPAD];
    const int tid = threadIdx.x;
    const int b   = blockIdx.x / (NN / 64);
    const int n0  = (blockIdx.x % (NN / 64)) * 64;
    stage_x_tile(ldsX, x, b, n0, tid);
    __syncthreads();

    const int lane = tid & 31, wave = tid >> 5;
    const int mi = lane & 15, half = lane >> 4;

    v8f acc[2][4] = {};
    for (int cc0 = 0; cc0 < NC; cc0 += 32) {
        V16U bx[4];
#pragma unroll
        for (int ns = 0; ns < 4; ++ns) {
            const _Float16* p = &ldsX[(ns * 16 + mi) * XPAD + cc0 + half * 8];
            bx[ns].h[0] = *(const v8h*)p;
            bx[ns].h[1] = *(const v8h*)(p + 16);
        }
#pragma unroll
        for (int t = 0; t < 2; ++t) {
            const int ct = wave + t * 8;
            const _Float16* wr = wvh + (long long)(ct * 16 + mi) * NC + cc0 + half * 8;
            V16U a;
            a.h[0] = *(const v8h*)wr;         // K = cc0 + half*8 .. +7
            a.h[1] = *(const v8h*)(wr + 16);  // K = cc0 + 16 + half*8 .. +7
#pragma unroll
            for (int ns = 0; ns < 4; ++ns)
                acc[t][ns] = wmma_f16(a.v, bx[ns].v, acc[t][ns]);
        }
    }
#pragma unroll
    for (int t = 0; t < 2; ++t) {
        const int ct = wave + t * 8;
#pragma unroll
        for (int r = 0; r < 8; ++r) {
            const int c = ct * 16 + 8 * half + r;
            const float bb = bv[c];
#pragma unroll
            for (int ns = 0; ns < 4; ++ns)
                vh[((long long)b * NC + c) * NN + n0 + ns * 16 + mi] =
                    (_Float16)(acc[t][ns][r] + bb);
        }
    }
}

// ---------------------------------------------------------------------------
// Q/K projection as WMMA GEMM, output f16 [B,N,CQ].
// block = 256 thr (8 waves) per (b, 64-n chunk).
// wave w: matrix = (w>>2 ? K : Q), cq-tile = (w>>1)&1, n-subtiles {2(w&1), +1}.
// ---------------------------------------------------------------------------
__global__ __launch_bounds__(256) void proj_qk_kernel(
    const float* __restrict__ x,
    const _Float16* __restrict__ wqh, const float* __restrict__ bq,
    const _Float16* __restrict__ wkh, const float* __restrict__ bk,
    _Float16* __restrict__ qh, _Float16* __restrict__ kh)
{
    __shared__ __align__(16) _Float16 ldsX[64 * XPAD];
    const int tid = threadIdx.x;
    const int b   = blockIdx.x / (NN / 64);
    const int n0  = (blockIdx.x % (NN / 64)) * 64;
    stage_x_tile(ldsX, x, b, n0, tid);
    __syncthreads();

    const int lane = tid & 31, wave = tid >> 5;
    const int mi = lane & 15, half = lane >> 4;
    const int sel = wave >> 2;
    const int ct  = (wave >> 1) & 1;
    const int nsb = (wave & 1) * 2;
    const _Float16* wm = sel ? wkh : wqh;
    const float*    bm = sel ? bk : bq;
    _Float16*       oh = sel ? kh : qh;

    v8f acc[2] = {};
    for (int cc0 = 0; cc0 < NC; cc0 += 32) {
        const _Float16* wr = wm + (long long)(ct * 16 + mi) * NC + cc0 + half * 8;
        V16U a;
        a.h[0] = *(const v8h*)wr;
        a.h[1] = *(const v8h*)(wr + 16);
#pragma unroll
        for (int s = 0; s < 2; ++s) {
            const _Float16* p = &ldsX[((nsb + s) * 16 + mi) * XPAD + cc0 + half * 8];
            V16U bx;
            bx.h[0] = *(const v8h*)p;
            bx.h[1] = *(const v8h*)(p + 16);
            acc[s] = wmma_f16(a.v, bx.v, acc[s]);
        }
    }
#pragma unroll
    for (int r = 0; r < 8; ++r) {
        const int cq = ct * 16 + 8 * half + r;
        const float bb = bm[cq];
#pragma unroll
        for (int s = 0; s < 2; ++s) {
            const int n = n0 + (nsb + s) * 16 + mi;
            oh[((long long)b * NN + n) * NCQ + cq] = (_Float16)(acc[s][r] + bb);
        }
    }
}

// ---------------------------------------------------------------------------
// Fused flash attention. block = 128 thr (4 waves) per (b, 16-row i-tile).
// Wave w owns channels [64w, 64w+64) -> 4 PV accumulators.
// Per 32-j chunk, per wave: 2 QK WMMAs, DPP row-max, exp, P->LDS transpose,
// 1 ones-WMMA (row sums), 4 PV WMMAs. No cross-wave sync in the loop.
// ---------------------------------------------------------------------------
__global__ __launch_bounds__(128) void attn_kernel(
    const _Float16* __restrict__ qh,   // [B,N,CQ]
    const _Float16* __restrict__ kh,   // [B,N,CQ]  (K^T layout)
    const _Float16* __restrict__ vh,   // [B,C,N]
    const float*    __restrict__ x,
    const float*    __restrict__ gamma,
    float*          __restrict__ out)
{
    __shared__ __align__(16) _Float16 ldsP[4][16][32];

    const int tid  = threadIdx.x;
    const int lane = tid & 31, wave = tid >> 5;
    const int mi = lane & 15, half = lane >> 4;
    const int b  = blockIdx.x / (NN / 16);
    const int i0 = (blockIdx.x % (NN / 16)) * 16;
    const int c0 = wave * 64;

    const _Float16* qrow = qh + ((long long)b * NN + i0 + mi) * NCQ;
    V16U qa;
    qa.h[0] = *(const v8h*)(qrow + half * 8);
    qa.h[1] = *(const v8h*)(qrow + 16 + half * 8);

    v16h onev;
#pragma unroll
    for (int i = 0; i < 16; ++i) onev[i] = (_Float16)1.0f;

    v8f acc[4] = {};
    v8f accL = {};
    const v8f zc = {};
    float mrow[8];
#pragma unroll
    for (int r = 0; r < 8; ++r) mrow[r] = -3.0e38f;

    const _Float16* kbase = kh + ((long long)b * NN) * NCQ;
    const _Float16* vbase = vh + ((long long)b * NC) * NN;

    for (int j0 = 0; j0 < NN; j0 += 32) {
        V16U kb0, kb1;
        const _Float16* kr0 = kbase + (long long)(j0 + mi) * NCQ + half * 8;
        kb0.h[0] = *(const v8h*)kr0;
        kb0.h[1] = *(const v8h*)(kr0 + 16);
        const _Float16* kr1 = kr0 + 16 * NCQ;
        kb1.h[0] = *(const v8h*)kr1;
        kb1.h[1] = *(const v8h*)(kr1 + 16);

        v8f e0 = wmma_f16(qa.v, kb0.v, zc);
        v8f e1 = wmma_f16(qa.v, kb1.v, zc);

        float p0[8], p1[8];
#pragma unroll
        for (int r = 0; r < 8; ++r) {
            const float t  = rowmax16(fmaxf(e0[r], e1[r]));
            const float mn = fmaxf(mrow[r], t);
            const float s  = __expf(mrow[r] - mn);
            mrow[r] = mn;
            p0[r] = __expf(e0[r] - mn);
            p1[r] = __expf(e1[r] - mn);
            accL[r]   *= s;
            acc[0][r] *= s;
            acc[1][r] *= s;
            acc[2][r] *= s;
            acc[3][r] *= s;
        }
#pragma unroll
        for (int r = 0; r < 8; ++r) {
            const int row = 8 * half + r;
            ldsP[wave][row][mi]      = (_Float16)p0[r];
            ldsP[wave][row][16 + mi] = (_Float16)p1[r];
        }
        asm volatile("" ::: "memory");  // LDS is in-order per wave; keep store->load order

        V16U pa;
        pa.h[0] = *(const v8h*)&ldsP[wave][mi][half * 8];
        pa.h[1] = *(const v8h*)&ldsP[wave][mi][16 + half * 8];

        accL = wmma_f16(pa.v, onev, accL);   // row sums via matrix pipe
#pragma unroll
        for (int t = 0; t < 4; ++t) {
            const _Float16* vr = vbase + (long long)(c0 + t * 16 + mi) * NN + j0 + half * 8;
            V16U vb;
            vb.h[0] = *(const v8h*)vr;
            vb.h[1] = *(const v8h*)(vr + 16);
            acc[t] = wmma_f16(pa.v, vb.v, acc[t]);
        }
    }

    const float g = gamma[0];
#pragma unroll
    for (int r = 0; r < 8; ++r) {
        const int row = 8 * half + r;
        const float iv = 1.0f / accL[r];
#pragma unroll
        for (int t = 0; t < 4; ++t) {
            const long long o = ((long long)b * NC + c0 + t * 16 + mi) * NN + i0 + row;
            out[o] = fmaf(g, acc[t][r] * iv, x[o]);
        }
    }
}

// ---------------------------------------------------------------------------
extern "C" void kernel_launch(void* const* d_in, const int* in_sizes, int n_in,
                              void* d_out, int out_size, void* d_ws, size_t ws_size,
                              hipStream_t stream)
{
    const float* x     = (const float*)d_in[0];
    const float* wq    = (const float*)d_in[1];
    const float* bq    = (const float*)d_in[2];
    const float* wk    = (const float*)d_in[3];
    const float* bk    = (const float*)d_in[4];
    const float* wv    = (const float*)d_in[5];
    const float* bv    = (const float*)d_in[6];
    const float* gamma = (const float*)d_in[7];
    float* out = (float*)d_out;

    _Float16* qhp = (_Float16*)d_ws;                       // 2 MB
    _Float16* khp = qhp + (size_t)NB * NN * NCQ;           // 2 MB
    _Float16* vhp = khp + (size_t)NB * NN * NCQ;           // 16 MB
    _Float16* wvh = vhp + (size_t)NB * NC * NN;            // 128 KB
    _Float16* wqh = wvh + (size_t)NC * NC;                 // 16 KB
    _Float16* wkh = wqh + (size_t)NCQ * NC;                // 16 KB

    cvt_w_kernel<<<(NC * NC + 255) / 256, 256, 0, stream>>>(wv, wq, wk, wvh, wqh, wkh);
    proj_v_kernel<<<NB * (NN / 64), 256, 0, stream>>>(x, wvh, bv, vhp);
    proj_qk_kernel<<<NB * (NN / 64), 256, 0, stream>>>(x, wqh, bq, wkh, bk, qhp, khp);
    attn_kernel<<<NB * (NN / 16), 128, 0, stream>>>(qhp, khp, vhp, x, gamma, out);
}
